// GraphStateLSTM_65317862637632
// MI455X (gfx1250) — matrix operand order repaired and checked by reading, not compiled, not run on hardware
//
#include <hip/hip_runtime.h>
#include <math.h>

#define B_   8
#define N_   2048
#define H_   512
#define M_   96
#define S_   16
#define E_   256
#define D_   64
#define R_   5
#define MAXD_ 600
#define NEG_INF_ (-1e10f)
#define NSPLIT 8
#define APITCH 520   // 8-half pad per row: 1040B row pitch -> bank-conflict-free b128 LDS reads

typedef __attribute__((ext_vector_type(16))) _Float16 v16h;
typedef __attribute__((ext_vector_type(8)))  _Float16 v8h;
typedef __attribute__((ext_vector_type(8)))  float    v8f;

// ---------------------------------------------------------------------------
// 0) one-shot: convert W_chem / W_dis [H,E] f32 -> transposed f16 Wt[2][E][H]
//    (K contiguous per output column -> B-fragments become two b128 loads)
// ---------------------------------------------------------------------------
__global__ __launch_bounds__(256) void k_wcvt(const float* __restrict__ Wc,
                                              const float* __restrict__ Wd,
                                              _Float16* __restrict__ Wt) {
  int idx = blockIdx.x * 256 + threadIdx.x;      // 2*E*H = 262144
  if (idx >= 2 * E_ * H_) return;
  int which = idx / (E_ * H_);
  int rem = idx % (E_ * H_);
  int n = rem / H_, k = rem % H_;                // contiguous writes over k
  const float* src = which ? Wd : Wc;
  Wt[idx] = (_Float16)src[(size_t)k * E_ + n];
}

// ---------------------------------------------------------------------------
// 1) sentence max-pool, split over N for parallelism
// ---------------------------------------------------------------------------
__global__ __launch_bounds__(128) void k_smax_part(const float* __restrict__ nh,
                                                   float* __restrict__ part) {
  int h = blockIdx.x * 128 + threadIdx.x;        // 4 blocks * 128 = 512
  int b = blockIdx.y;
  int z = blockIdx.z;
  const int chunk = N_ / NSPLIT;                 // 256
  const float* p = nh + ((size_t)b * N_ + (size_t)z * chunk) * H_ + h;
  float m = -INFINITY;
  for (int n = 0; n < chunk; ++n) m = fmaxf(m, p[(size_t)n * H_]);
  part[((size_t)b * NSPLIT + z) * H_ + h] = m;
}

__global__ __launch_bounds__(256) void k_smax_red(const float* __restrict__ part,
                                                  float* __restrict__ sent) {
  int idx = blockIdx.x * 256 + threadIdx.x;      // 16 blocks -> 4096
  int b = idx / H_, h = idx % H_;
  float m = -INFINITY;
  for (int z = 0; z < NSPLIT; ++z) m = fmaxf(m, part[((size_t)b * NSPLIT + z) * H_ + h]);
  sent[idx] = m;
}

// ---------------------------------------------------------------------------
// 2) entity path: gather spans -> masked sum -> GEMM(512x256) via WMMA -> tanh
//    one block = 16 mention-rows; 4 waves x 4 n-tiles cover E=256.
// ---------------------------------------------------------------------------
__global__ __launch_bounds__(128) void k_entity_gemm(
    const float* __restrict__ nh, const int* __restrict__ pmap,
    const float* __restrict__ pmask, const _Float16* __restrict__ Wt,
    const float* __restrict__ bias, float* __restrict__ out,
    float* __restrict__ valid) {
  __shared__ int      s_pos[16][16];
  __shared__ float    s_msk[16][16];
  __shared__ _Float16 Alds[16 * APITCH];

  const int tid = threadIdx.x;
  const int m0  = blockIdx.x * 16;               // global row base (96 % 16 == 0)

  for (int i = tid; i < 256; i += 128) {
    int row = m0 + (i >> 4), s = i & 15;
    s_pos[i >> 4][s] = pmap[row * S_ + s];
    s_msk[i >> 4][s] = pmask[row * S_ + s];
  }
  __syncthreads();

  if (tid < 16) {
    float sm = 0.f;
    for (int s = 0; s < S_; ++s) sm += s_msk[tid][s];
    valid[m0 + tid] = (sm > 0.f) ? 1.f : 0.f;
  }

  // gather + masked sum: 8 threads per mention row, float4 columns
  {
    const int mr = tid >> 3;
    const int hb = (tid & 7) * 4;
    const int row = m0 + mr;
    const int b = row / M_;
    const float* nbase = nh + (size_t)b * N_ * H_;
    for (int c = 0; c < 16; ++c) {
      int h = c * 32 + hb;
      float ax = 0.f, ay = 0.f, az = 0.f, aw = 0.f;
      for (int s = 0; s < S_; ++s) {
        float mk = s_msk[mr][s];
        const float4 v = *(const float4*)(nbase + (size_t)s_pos[mr][s] * H_ + h);
        ax += mk * v.x; ay += mk * v.y; az += mk * v.z; aw += mk * v.w;
      }
      _Float16* ap = Alds + mr * APITCH + h;
      ap[0] = (_Float16)ax; ap[1] = (_Float16)ay;
      ap[2] = (_Float16)az; ap[3] = (_Float16)aw;
    }
  }
  __syncthreads();

  // WMMA: wave w -> columns [w*64, w*64+64)
  const int wave = tid >> 5, lane = tid & 31;
  const int half = lane >> 4;                    // lane half
  const int l16  = lane & 15;
  const int nb0  = wave * 64;
  const int koff = half * 8;                     // A-frag K interleave
  const int kofB = half * 8;                     // B-frag K interleave

  v8f acc0 = {}, acc1 = {}, acc2 = {}, acc3 = {};

  for (int kt = 0; kt < 16; ++kt) {
    const int kb = kt * 32;
    // A fragment: lane<16 holds K kb+{0..7,16..23}; lane>=16: +8
    const _Float16* ap = Alds + l16 * APITCH + kb + koff;
    v8h alo = *(const v8h*)ap;
    v8h ahi = *(const v8h*)(ap + 16);
    v16h a = __builtin_shufflevector(alo, ahi, 0,1,2,3,4,5,6,7,8,9,10,11,12,13,14,15);

#pragma unroll
    for (int nt = 0; nt < 4; ++nt) {
      // B fragment: lanes 0-15 hold K kb+{0..7,16..23} of column n; lanes 16-31: +8
      const _Float16* wp = Wt + (size_t)(nb0 + nt * 16 + l16) * H_ + kb + kofB;
      v8h blo = *(const v8h*)wp;
      v8h bhi = *(const v8h*)(wp + 16);
      v16h bf = __builtin_shufflevector(blo, bhi, 0,1,2,3,4,5,6,7,8,9,10,11,12,13,14,15);
      v8f* accp = (nt == 0) ? &acc0 : (nt == 1) ? &acc1 : (nt == 2) ? &acc2 : &acc3;
      *accp = __builtin_amdgcn_wmma_f32_16x16x32_f16(
          false, a, false, bf, (short)0, *accp, false, false);
    }
  }

  // epilogue: bias + tanh; C/D layout: VGPR v -> row v (+8 for upper lanes)
#pragma unroll
  for (int nt = 0; nt < 4; ++nt) {
    const v8f acc = (nt == 0) ? acc0 : (nt == 1) ? acc1 : (nt == 2) ? acc2 : acc3;
    const int coln = nb0 + nt * 16 + l16;
    const float bv = bias[coln];
#pragma unroll
    for (int v = 0; v < 8; ++v) {
      const int rrow = v + half * 8;
      out[(size_t)(m0 + rrow) * E_ + coln] = tanhf(acc[v] + bv);
    }
  }
}

// ---------------------------------------------------------------------------
// 3) small projections
// ---------------------------------------------------------------------------
__global__ __launch_bounds__(256) void k_distT(const float* __restrict__ demb,
                                               const float* __restrict__ Ws,
                                               float* __restrict__ dT) {
  int idx = blockIdx.x * 256 + threadIdx.x;
  if (idx >= MAXD_ * R_) return;
  int d = idx / R_, r = idx % R_;
  float s = 0.f;
  for (int k = 0; k < D_; ++k) s += demb[d * D_ + k] * Ws[(2 * E_ + k) * R_ + r];
  dT[idx] = s;
}

__global__ __launch_bounds__(64) void k_sentS(const float* __restrict__ sent,
                                              const float* __restrict__ Ws,
                                              const float* __restrict__ bs,
                                              float* __restrict__ sS) {
  int t = threadIdx.x;
  if (t >= B_ * R_) return;
  int b = t / R_, r = t % R_;
  float s = bs[r];
  for (int h = 0; h < H_; ++h) s += sent[b * H_ + h] * Ws[(2 * E_ + D_ + h) * R_ + r];
  sS[t] = s;
}

__global__ __launch_bounds__(256) void k_proj(const float* __restrict__ chemE,
                                              const float* __restrict__ disE,
                                              const float* __restrict__ Ws,
                                              float* __restrict__ AB) {
  int idx = blockIdx.x * 256 + threadIdx.x;
  if (idx >= 2 * B_ * M_ * R_) return;
  int which = idx / (B_ * M_ * R_);
  int rem = idx % (B_ * M_ * R_);
  int bm = rem / R_, r = rem % R_;
  const float* ent = which ? disE : chemE;
  const int wbase = which * E_;
  float s = 0.f;
  for (int e = 0; e < E_; ++e) s += ent[(size_t)bm * E_ + e] * Ws[(wbase + e) * R_ + r];
  AB[idx] = s;
}

// ---------------------------------------------------------------------------
// 4) final masked max over all (i,j) pairs; everything staged in LDS
// ---------------------------------------------------------------------------
__global__ __launch_bounds__(256) void k_final(const int* __restrict__ distance,
                                               const float* __restrict__ AB,
                                               const float* __restrict__ dT,
                                               const float* __restrict__ sS,
                                               const float* __restrict__ cval,
                                               const float* __restrict__ dval,
                                               float* __restrict__ out) {
  __shared__ float sA[M_ * R_], sB[M_ * R_], sT[MAXD_ * R_], sSent[8];
  __shared__ float cv[M_], dv[M_];
  __shared__ float red[256 * R_];
  const int b = blockIdx.x, tid = threadIdx.x;
  const float* As = AB + (size_t)b * M_ * R_;
  const float* Bs = AB + (size_t)(B_ + b) * M_ * R_;

  for (int i = tid; i < M_ * R_; i += 256) { sA[i] = As[i]; sB[i] = Bs[i]; }
  for (int i = tid; i < MAXD_ * R_; i += 256) sT[i] = dT[i];
  if (tid < R_) sSent[tid] = sS[b * R_ + tid];
  for (int i = tid; i < M_; i += 256) { cv[i] = cval[b * M_ + i]; dv[i] = dval[b * M_ + i]; }
  __syncthreads();

  float mx[R_];
#pragma unroll
  for (int r = 0; r < R_; ++r) mx[r] = NEG_INF_;

  for (int p = tid; p < M_ * M_; p += 256) {
    int i = p / M_, j = p % M_;
    if (cv[i] > 0.f && dv[j] > 0.f) {
      int dd = distance[((size_t)b * M_ + i) * M_ + j];
      if (dd >= 0) {
#pragma unroll
        for (int r = 0; r < R_; ++r) {
          float v = sA[i * R_ + r] + sB[j * R_ + r] + sT[dd * R_ + r] + sSent[r];
          mx[r] = fmaxf(mx[r], v);
        }
      }
    }
  }
#pragma unroll
  for (int r = 0; r < R_; ++r) red[tid * R_ + r] = mx[r];
  __syncthreads();
  for (int s = 128; s > 0; s >>= 1) {
    if (tid < s)
#pragma unroll
      for (int r = 0; r < R_; ++r)
        red[tid * R_ + r] = fmaxf(red[tid * R_ + r], red[(tid + s) * R_ + r]);
    __syncthreads();
  }
  if (tid < R_) out[b * R_ + tid] = red[tid];
}

// ---------------------------------------------------------------------------
extern "C" void kernel_launch(void* const* d_in, const int* in_sizes, int n_in,
                              void* d_out, int out_size, void* d_ws, size_t ws_size,
                              hipStream_t stream) {
  const float* nh    = (const float*)d_in[0];
  const int*   cmap  = (const int*)  d_in[1];
  const float* cmask = (const float*)d_in[2];
  const int*   dmap  = (const int*)  d_in[3];
  const float* dmask = (const float*)d_in[4];
  const int*   dist  = (const int*)  d_in[5];
  const float* Wc    = (const float*)d_in[6];
  const float* bc    = (const float*)d_in[7];
  const float* Wd    = (const float*)d_in[8];
  const float* bd    = (const float*)d_in[9];
  const float* Ws    = (const float*)d_in[10];
  const float* bs    = (const float*)d_in[11];
  const float* demb  = (const float*)d_in[12];
  float* out = (float*)d_out;
  float* ws  = (float*)d_ws;

  float* part  = ws;                          // 8*8*512      = 32768
  float* sent  = ws + 32768;                  // 8*512        = 4096
  float* chemE = ws + 36864;                  // 768*256      = 196608
  float* disE  = ws + 233472;                 // 768*256      = 196608
  float* cval  = ws + 430080;                 // 768
  float* dval  = ws + 430848;                 // 768
  float* AB    = ws + 431616;                 // 2*768*5      = 7680
  float* dT    = ws + 439296;                 // 600*5        = 3000
  float* sS    = ws + 442296;                 // 40 (pad to 442368 for alignment)
  _Float16* Wt = (_Float16*)(ws + 442368);    // 2*256*512 halfs = 512 KB

  k_wcvt<<<(2 * E_ * H_ + 255) / 256, 256, 0, stream>>>(Wc, Wd, Wt);

  k_smax_part<<<dim3(H_ / 128, B_, NSPLIT), 128, 0, stream>>>(nh, part);
  k_smax_red<<<(B_ * H_) / 256, 256, 0, stream>>>(part, sent);

  k_entity_gemm<<<(B_ * M_) / 16, 128, 0, stream>>>(nh, cmap, cmask, Wt, bc, chemE, cval);
  k_entity_gemm<<<(B_ * M_) / 16, 128, 0, stream>>>(nh, dmap, dmask, Wt + (size_t)E_ * H_,
                                                    bd, disE, dval);

  k_distT<<<(MAXD_ * R_ + 255) / 256, 256, 0, stream>>>(demb, Ws, dT);
  k_sentS<<<1, 64, 0, stream>>>(sent, Ws, bs, sS);
  k_proj<<<(2 * B_ * M_ * R_ + 255) / 256, 256, 0, stream>>>(chemE, disE, Ws, AB);

  k_final<<<B_, 256, 0, stream>>>(dist, AB, dT, sS, cval, dval, out);
}
